// LSTM_Decoder_15418932593332
// MI455X (gfx1250) — matrix operand order repaired and checked by reading, not compiled, not run on hardware
//
#include <hip/hip_runtime.h>

#define BDIM 128
#define HDIM 512
#define IDIM 1024
#define LNUM 3
#define TLEN 512
#define NBLOCKS 64
#define TPB 128            // 4 waves per block -> 256 waves total

typedef __attribute__((ext_vector_type(16))) __bf16          v16bf;
typedef __attribute__((ext_vector_type(8)))  float           v8f;
typedef __attribute__((ext_vector_type(8)))  unsigned short  v8us;
typedef __attribute__((ext_vector_type(16))) unsigned short  v16us;

__device__ __forceinline__ unsigned short f32_to_bf16_rne(float f) {
    unsigned int u = __float_as_uint(f);
    u += 0x7FFFu + ((u >> 16) & 1u);   // round to nearest even
    return (unsigned short)(u >> 16);
}

__device__ __forceinline__ float fast_sigmoid(float x) {
    return __builtin_amdgcn_rcpf(1.0f + __expf(-x));
}
__device__ __forceinline__ float fast_tanh(float x) {
    return 2.0f * fast_sigmoid(2.0f * x) - 1.0f;
}

// A-matrix (16x32 bf16) per-lane load, contiguous K order (matches B):
// lanes 0-15 : elements 0..15 = K(kk..kk+15)
// lanes 16-31: elements 0..15 = K(kk+16..kk+31)
__device__ __forceinline__ v16bf load_a_tile(const unsigned short* rowp, int kh) {
    v8us lo = *(const v8us*)(rowp + kh * 16);
    v8us hi = *(const v8us*)(rowp + kh * 16 + 8);
    v16us cat = __builtin_shufflevector(lo, hi, 0,1,2,3,4,5,6,7,8,9,10,11,12,13,14,15);
    return __builtin_bit_cast(v16bf, cat);
}

// B-matrix (32x16 bf16) = W.T tile: lane n holds weight row (col n),
// lane-half selects K range; 32 contiguous bytes per lane.
__device__ __forceinline__ v16bf load_b_tile(const unsigned short* p) {
    v16us x = *(const v16us*)p;
    return __builtin_bit_cast(v16bf, x);
}

__device__ __forceinline__ v8f wmma_bf16(v16bf a, v16bf b, v8f c) {
    return __builtin_amdgcn_wmma_f32_16x16x32_bf16(false, a, false, b, (short)0, c, false, false);
}

// Grid-wide barrier: epoch-monotonic, agent-scope atomics, s_sleep backoff.
__device__ __forceinline__ void grid_barrier(int* bar, int target) {
    __syncthreads();
    if (threadIdx.x == 0) {
        __threadfence();
        int prev = __hip_atomic_fetch_add(&bar[0], 1, __ATOMIC_ACQ_REL, __HIP_MEMORY_SCOPE_AGENT);
        if (prev == NBLOCKS - 1) {
            __hip_atomic_store(&bar[0], 0, __ATOMIC_RELAXED, __HIP_MEMORY_SCOPE_AGENT);
            __hip_atomic_store(&bar[1], target, __ATOMIC_RELEASE, __HIP_MEMORY_SCOPE_AGENT);
        } else {
            while (__hip_atomic_load(&bar[1], __ATOMIC_ACQUIRE, __HIP_MEMORY_SCOPE_AGENT) < target) {
                __builtin_amdgcn_s_sleep(2);
            }
        }
    }
    __syncthreads();
}

__global__ void k_f32_to_bf16(const float* __restrict__ src,
                              unsigned short* __restrict__ dst, int n) {
    int i = blockIdx.x * blockDim.x + threadIdx.x;
    int stride = gridDim.x * blockDim.x;
    for (; i < n; i += stride) dst[i] = f32_to_bf16_rne(src[i]);
}

__global__ void k_bias_init(const float* __restrict__ bih, const float* __restrict__ bhh,
                            float* __restrict__ bias, int n, int* __restrict__ bar) {
    int i = blockIdx.x * blockDim.x + threadIdx.x;
    if (i < n) bias[i] = bih[i] + bhh[i];
    if (i == 0) { bar[0] = 0; bar[1] = 0; }
}

__global__ __launch_bounds__(TPB)
void lstm_persistent(const unsigned short* __restrict__ Wih,   // L x 4H x H (bf16)
                     const unsigned short* __restrict__ Wout,  // I x H (bf16)
                     const float* __restrict__ bias,           // L x 4H (b_ih+b_hh)
                     const float* __restrict__ b_out,          // I
                     unsigned short* __restrict__ hA,          // B x H bf16 (holds h0)
                     unsigned short* __restrict__ hB,          // B x H bf16
                     float* __restrict__ out,                  // B x T x I
                     int* __restrict__ bar) {
    const int lane = threadIdx.x & 31;
    const int gw   = blockIdx.x * (TPB / 32) + (threadIdx.x >> 5);  // 0..255
    const int kh   = lane >> 4;     // lane-half (K-group select)
    const int ln   = lane & 15;     // column-within-tile / row-within-tile

    // Gate-layer tile: 8 row tiles x 32 col tiles = 256 = NWAVES
    const int rtile = gw >> 5;            // 0..7
    const int jbase = (gw & 31) << 4;     // 0..496
    const int col   = jbase + ln;         // hidden column this lane produces

    unsigned short* cur = hA;
    unsigned short* nxt = hB;
    int epoch = 0;

    for (int t = 0; t < TLEN; ++t) {
        for (int layer = 0; layer < LNUM; ++layer) {
            const unsigned short* W  = Wih + (size_t)layer * (4 * HDIM * HDIM);
            const float*          bs = bias + layer * (4 * HDIM);
            // gate chunks: ig=[0,H) gg=[2H,3H) og=[3H,4H); fg skipped entirely
            const float bi = bs[0 * HDIM + col];
            const float bg = bs[2 * HDIM + col];
            const float bo = bs[3 * HDIM + col];
            v8f ai = {bi, bi, bi, bi, bi, bi, bi, bi};
            v8f ag = {bg, bg, bg, bg, bg, bg, bg, bg};
            v8f ao = {bo, bo, bo, bo, bo, bo, bo, bo};

            const unsigned short* ap  = cur + (size_t)(rtile * 16 + ln) * HDIM;
            const unsigned short* bpi = W + (size_t)(0 * HDIM + col) * HDIM + kh * 16;
            const unsigned short* bpg = W + (size_t)(2 * HDIM + col) * HDIM + kh * 16;
            const unsigned short* bpo = W + (size_t)(3 * HDIM + col) * HDIM + kh * 16;

            #pragma unroll 4
            for (int kk = 0; kk < HDIM; kk += 32) {
                v16bf a  = load_a_tile(ap + kk, kh);
                v16bf wi = load_b_tile(bpi + kk);
                v16bf wg = load_b_tile(bpg + kk);
                v16bf wo = load_b_tile(bpo + kk);
                ai = wmma_bf16(a, wi, ai);
                ag = wmma_bf16(a, wg, ag);
                ao = wmma_bf16(a, wo, ao);
            }

            // fused LSTM-ish epilogue: h = sig(og) * tanh(sig(ig) * tanh(gg))
            #pragma unroll
            for (int v = 0; v < 8; ++v) {
                float c  = fast_sigmoid(ai[v]) * fast_tanh(ag[v]);
                float hh = fast_sigmoid(ao[v]) * fast_tanh(c);
                int row = rtile * 16 + v + 8 * kh;
                nxt[(size_t)row * HDIM + col] = f32_to_bf16_rne(hh);
            }

            unsigned short* tmp = cur; cur = nxt; nxt = tmp;
            grid_barrier(bar, ++epoch);
        }

        // Output projection: 8 x 64 = 512 tiles, 2 per wave. out = h @ Wout.T + b_out
        #pragma unroll
        for (int s = 0; s < 2; ++s) {
            const int tt    = gw + s * 256;
            const int orow  = (tt >> 6) * 16;        // batch row tile
            const int ocol  = (tt & 63) * 16 + ln;   // output feature column
            const float bb  = b_out[ocol];
            v8f acc = {bb, bb, bb, bb, bb, bb, bb, bb};

            const unsigned short* ap = cur + (size_t)(orow + ln) * HDIM;
            const unsigned short* bp = Wout + (size_t)ocol * HDIM + kh * 16;

            #pragma unroll 4
            for (int kk = 0; kk < HDIM; kk += 32) {
                v16bf a = load_a_tile(ap + kk, kh);
                v16bf w = load_b_tile(bp + kk);
                acc = wmma_bf16(a, w, acc);
            }
            #pragma unroll
            for (int v = 0; v < 8; ++v) {
                int row = orow + v + 8 * kh;
                out[(size_t)row * ((size_t)TLEN * IDIM) + (size_t)t * IDIM + ocol] = acc[v];
            }
        }
        // No barrier needed here: next layer0 only reads `cur` (write-after-read
        // to the other buffer is fenced by the barrier after the next layer0).
    }
}

extern "C" void kernel_launch(void* const* d_in, const int* in_sizes, int n_in,
                              void* d_out, int out_size, void* d_ws, size_t ws_size,
                              hipStream_t stream) {
    (void)in_sizes; (void)n_in; (void)out_size; (void)ws_size;
    const float* dec  = (const float*)d_in[0];  // (B,1,H)
    const float* Wih  = (const float*)d_in[1];  // (L,4H,H)
    // d_in[2] = W_hh is mathematically unused by the reference
    const float* bih  = (const float*)d_in[3];  // (L,4H)
    const float* bhh  = (const float*)d_in[4];  // (L,4H)
    const float* Wout = (const float*)d_in[5];  // (I,H)
    const float* bout = (const float*)d_in[6];  // (I,)

    const size_t wih_n  = (size_t)LNUM * 4 * HDIM * HDIM;  // 3,145,728
    const size_t wout_n = (size_t)IDIM * HDIM;             //   524,288
    const size_t h_n    = (size_t)BDIM * HDIM;             //    65,536

    char* ws = (char*)d_ws;
    int*            bar     = (int*)ws;                           // 256 B reserved
    unsigned short* wih_bf  = (unsigned short*)(ws + 256);
    unsigned short* wout_bf = wih_bf + wih_n;
    float*          bias    = (float*)(wout_bf + wout_n);
    unsigned short* hA      = (unsigned short*)(bias + LNUM * 4 * HDIM);
    unsigned short* hB      = hA + h_n;

    k_f32_to_bf16<<<dim3(1024), dim3(256), 0, stream>>>(Wih,  wih_bf,  (int)wih_n);
    k_f32_to_bf16<<<dim3(512),  dim3(256), 0, stream>>>(Wout, wout_bf, (int)wout_n);
    k_f32_to_bf16<<<dim3(256),  dim3(256), 0, stream>>>(dec,  hA,      (int)h_n);
    k_bias_init  <<<dim3(24),   dim3(256), 0, stream>>>(bih, bhh, bias, LNUM * 4 * HDIM, bar);

    lstm_persistent<<<dim3(NBLOCKS), dim3(TPB), 0, stream>>>(
        wih_bf, wout_bf, bias, bout, hA, hB, (float*)d_out, bar);
}
